// CausalWanS2VSelfAttention_5875515261494
// MI455X (gfx1250) — compile-verified, matrix operand order checked by reading, not run on hardware
//
#include <hip/hip_runtime.h>
#include <hip/hip_bf16.h>

// ---------------------------------------------------------------------------
// CDNA5 (gfx1250, wave32) implementation of CausalWanS2VSelfAttention.
// Matrix math: v_wmma_f32_16x16x32_bf16.  KV staging: Tensor Data Mover
// (tensor_load_to_lds, TENSORcnt) with double-buffered LDS tiles.
// ---------------------------------------------------------------------------

typedef __attribute__((ext_vector_type(16))) __bf16 v16bf;
typedef __attribute__((ext_vector_type(8)))  __bf16 v8bf;
typedef __attribute__((ext_vector_type(8)))  float  v8f;
typedef __attribute__((ext_vector_type(4)))  unsigned int u32x4;
typedef __attribute__((ext_vector_type(8)))  int    i32x8;
typedef __attribute__((ext_vector_type(4)))  int    i32x4;

union Frag {
    v16bf f;
    v8bf  h[2];
    __bf16 e[16];
};
union Out8 {
    v8bf v;
    __bf16 e[8];
};

__device__ __forceinline__ v8f wmma_bf16(v16bf a, v16bf b, v8f c) {
    // 8 args: (neg_a, A, neg_b, B, c_mod, C, reuse_a, reuse_b)
    return __builtin_amdgcn_wmma_f32_16x16x32_bf16(false, a, false, b, (short)0, c,
                                                   false, false);
}

#define SEQ        720
#define DIM        2048
#define N_HEADS    16
#define HEAD_DIM   128
#define CACHE_SIZE 11520

// ---------------------------------------------------------------------------
// TDM: 2-D tile load (bf16 elements) global -> LDS via Tensor DMA descriptor.
// Tile rows are packed densely into LDS (row-major, no padding).
//   lds_addr     : byte offset of destination in LDS
//   gptr         : global address of tile start
//   tile_d0/d1   : tile width (elements) / height (rows)
//   tensor_d0/d1 : tensor extents for OOB checking (elements / rows)
//   d0_stride    : row stride in elements
// This toolchain exposes the 6-arg builtin:
//   (u32x4 g0, i32x8 g1, i32x4 g2, i32x4 g3, i32x8 extra, i32 cpol)
// ---------------------------------------------------------------------------
__device__ __forceinline__ void tdm_load_2d_bf16(unsigned int lds_addr, const void* gptr,
                                                 unsigned int tile_d0, unsigned int tile_d1,
                                                 unsigned int tensor_d0, unsigned int tensor_d1,
                                                 unsigned long long d0_stride) {
    unsigned long long ga = (unsigned long long)gptr;
    u32x4 g0;
    g0[0] = 1u;                                            // count=1, user descriptor
    g0[1] = lds_addr;                                      // LDS byte address
    g0[2] = (unsigned int)ga;                              // global_addr[31:0]
    g0[3] = (unsigned int)((ga >> 32) & 0x1FFFFFFull)      // global_addr[56:32]
          | (2u << 30);                                    // type = 2 ("image")
    i32x8 g1;
    g1[0] = (int)(1u << 16);                               // data_size=1 -> 2 bytes
    g1[1] = (int)((tensor_d0 & 0xFFFFu) << 16);            // tensor_dim0[15:0]
    g1[2] = (int)(((tensor_d0 >> 16) & 0xFFFFu)            // tensor_dim0[31:16]
          |       ((tensor_d1 & 0xFFFFu) << 16));          // tensor_dim1[15:0]
    g1[3] = (int)(((tensor_d1 >> 16) & 0xFFFFu)            // tensor_dim1[31:16]
          |       ((tile_d0 & 0xFFFFu) << 16));            // tile_dim0
    g1[4] = (int)(tile_d1 & 0xFFFFu);                      // tile_dim1 (tile_dim2=0)
    g1[5] = (int)(unsigned int)(d0_stride & 0xFFFFFFFFull);// tensor_dim0_stride[31:0]
    g1[6] = (int)(unsigned int)((d0_stride >> 32) & 0xFFFFull); // stride[47:32]
    g1[7] = 0;                                             // tensor_dim1_stride = 0
    i32x4 z4 = {0, 0, 0, 0};
    i32x8 z8 = {0, 0, 0, 0, 0, 0, 0, 0};
    __builtin_amdgcn_tensor_load_to_lds(g0, g1, z4, z4, z8, 0);
}

// ---------------------------------------------------------------------------
// f32 -> bf16 elementwise convert
// ---------------------------------------------------------------------------
__global__ void k_f32_to_bf16(const float* __restrict__ in, __bf16* __restrict__ out, int n) {
    int i = blockIdx.x * blockDim.x + threadIdx.x;
    if (i < n) out[i] = (__bf16)in[i];
}

// ---------------------------------------------------------------------------
// cache_k [n][h][d] f32 -> k_all [h][n][d] bf16
// cache_v [n][h][d] f32 -> vT    [h][d][n] bf16
// ---------------------------------------------------------------------------
__global__ void k_cache_convert(const float* __restrict__ ck, const float* __restrict__ cv,
                                __bf16* __restrict__ kall, __bf16* __restrict__ vT) {
    int i = blockIdx.x * blockDim.x + threadIdx.x;
    const int total = CACHE_SIZE * N_HEADS * HEAD_DIM;
    if (i >= total) return;
    int n = i / (N_HEADS * HEAD_DIM);
    int c = i - n * (N_HEADS * HEAD_DIM);
    int h = c >> 7;
    int d = c & 127;
    kall[((size_t)h * CACHE_SIZE + n) * HEAD_DIM + d] = (__bf16)ck[i];
    vT[((size_t)h * HEAD_DIM + d) * CACHE_SIZE + n]   = (__bf16)cv[i];
}

// ---------------------------------------------------------------------------
// Generic NT GEMM: C[M][N] = A[M][K] * B[N][K]^T + bias[N]
// A, B bf16 row-major (K contiguous), C f32 row-major.
// One wave computes a 16x64 tile (1 A-frag reused over 4 B-frags).
// ---------------------------------------------------------------------------
__global__ __launch_bounds__(256) void k_gemm_nt(const __bf16* __restrict__ A,
                                                 const __bf16* __restrict__ B,
                                                 const float* __restrict__ bias,
                                                 float* __restrict__ C,
                                                 int M, int N, int K, int tilesN) {
    const int lane  = threadIdx.x & 31;
    const int wave  = threadIdx.x >> 5;
    const int gwave = blockIdx.x * 8 + wave;
    const int mt  = gwave / tilesN;          // 16-row tile
    const int ntg = gwave - mt * tilesN;     // 64-col group
    if (mt * 16 >= M) return;

    const int l16 = lane & 15;
    const int kh  = (lane >> 4) << 3;        // 0 or 8: K-half per A/B layout

    const __bf16* arow = A + (size_t)(mt * 16 + l16) * K;
    v8f acc0 = {}, acc1 = {}, acc2 = {}, acc3 = {};

    for (int k0 = 0; k0 < K; k0 += 32) {
        Frag a;
        a.h[0] = *(const v8bf*)(arow + k0 + kh);
        a.h[1] = *(const v8bf*)(arow + k0 + kh + 16);

        const __bf16* b0 = B + (size_t)((ntg * 4 + 0) * 16 + l16) * K + k0 + kh;
        const __bf16* b1 = B + (size_t)((ntg * 4 + 1) * 16 + l16) * K + k0 + kh;
        const __bf16* b2 = B + (size_t)((ntg * 4 + 2) * 16 + l16) * K + k0 + kh;
        const __bf16* b3 = B + (size_t)((ntg * 4 + 3) * 16 + l16) * K + k0 + kh;
        // global_prefetch a cacheline ahead of the streamed B rows
        __builtin_prefetch(b0 + 512, 0, 1);
        __builtin_prefetch(b1 + 512, 0, 1);
        Frag fb0, fb1, fb2, fb3;
        fb0.h[0] = *(const v8bf*)b0;  fb0.h[1] = *(const v8bf*)(b0 + 16);
        fb1.h[0] = *(const v8bf*)b1;  fb1.h[1] = *(const v8bf*)(b1 + 16);
        fb2.h[0] = *(const v8bf*)b2;  fb2.h[1] = *(const v8bf*)(b2 + 16);
        fb3.h[0] = *(const v8bf*)b3;  fb3.h[1] = *(const v8bf*)(b3 + 16);

        acc0 = wmma_bf16(a.f, fb0.f, acc0);
        acc1 = wmma_bf16(a.f, fb1.f, acc1);
        acc2 = wmma_bf16(a.f, fb2.f, acc2);
        acc3 = wmma_bf16(a.f, fb3.f, acc3);
    }

    // C layout: lane L, vgpr r -> (m = mt*16 + r + 8*(L>=16), n = ntile*16 + (L&15))
    const int mbase = mt * 16 + ((lane >> 4) << 3);
    v8f* accs[4] = { &acc0, &acc1, &acc2, &acc3 };
#pragma unroll
    for (int j = 0; j < 4; ++j) {
        const int n = (ntg * 4 + j) * 16 + l16;
        const float bs = bias[n];
        v8f a = *accs[j];
#pragma unroll
        for (int r = 0; r < 8; ++r) {
            C[(size_t)(mbase + r) * N + n] = a[r] + bs;
        }
    }
}

// ---------------------------------------------------------------------------
// RMSNorm (over DIM) + RoPE epilogue, bf16 output.
// dst layout: [h][row][128]; row = row_off + m, row_off from
// (cur - sink) % (CACHE-sink) + sink (or 0 when cur == nullptr).
// out_scale folds the 1/sqrt(HEAD_DIM) softmax scale into Q.
// ---------------------------------------------------------------------------
__global__ __launch_bounds__(256) void k_norm_rope(const float* __restrict__ y,
                                                   const float* __restrict__ nw,
                                                   const float* __restrict__ fcos,
                                                   const float* __restrict__ fsin,
                                                   __bf16* __restrict__ dst,
                                                   int rows_per_head,
                                                   const int* __restrict__ cur,
                                                   const int* __restrict__ sink,
                                                   float out_scale) {
    const int m = blockIdx.x;
    const float* row = y + (size_t)m * DIM;

    __shared__ float red[256];
    float ss = 0.f;
    for (int i = threadIdx.x; i < DIM; i += 256) {
        float v = row[i];
        ss += v * v;
    }
    red[threadIdx.x] = ss;
    __syncthreads();
    for (int s = 128; s > 0; s >>= 1) {
        if (threadIdx.x < s) red[threadIdx.x] += red[threadIdx.x + s];
        __syncthreads();
    }
    const float scale = rsqrtf(red[0] / (float)DIM + 1e-6f);

    int row_off = 0;
    if (cur != nullptr) {
        const int st = sink[0];
        row_off = (cur[0] - st) % (CACHE_SIZE - st) + st;
    }

    for (int p = threadIdx.x; p < DIM / 2; p += 256) {
        const int h = p >> 6;          // 64 pairs per head
        const int i = p & 63;
        const int d0 = h * HEAD_DIM + 2 * i;
        const float xr = row[d0]     * scale * nw[d0];
        const float xi = row[d0 + 1] * scale * nw[d0 + 1];
        const float c  = fcos[m * 64 + i];
        const float s  = fsin[m * 64 + i];
        const float o0 = (xr * c - xi * s) * out_scale;
        const float o1 = (xr * s + xi * c) * out_scale;
        __bf16* q = dst + ((size_t)h * rows_per_head + row_off + m) * HEAD_DIM;
        q[2 * i]     = (__bf16)o0;
        q[2 * i + 1] = (__bf16)o1;
    }
}

// ---------------------------------------------------------------------------
// V epilogue: yv [m][h*128+d] f32 -> vT [h][d][row_off + m] bf16 (transposed)
// ---------------------------------------------------------------------------
__global__ void k_v_transpose(const float* __restrict__ yv, __bf16* __restrict__ vT,
                              const int* __restrict__ cur, const int* __restrict__ sink) {
    int i = blockIdx.x * blockDim.x + threadIdx.x;
    const int total = SEQ * DIM;
    if (i >= total) return;
    const int st = sink[0];
    const int row_off = (cur[0] - st) % (CACHE_SIZE - st) + st;
    const int m = i / DIM;
    const int c = i - m * DIM;
    const int h = c >> 7;
    const int d = c & 127;
    vT[((size_t)h * HEAD_DIM + d) * CACHE_SIZE + row_off + m] = (__bf16)yv[i];
}

// ---------------------------------------------------------------------------
// Flash attention (full 11520-key window), bf16 WMMA, TDM-staged KV tiles.
// grid = (N_HEADS, 6); block = 256 threads = 8 waves; wave w owns queries
// qt = blockIdx.y*8 + w (16 queries each). Computes S^T = K*Q^T so that
// softmax stats are per-lane (query in lanes) and exp(S) feeds the PV WMMA
// B operand lane-locally. Output acc is out^T (d in VGPRs, query in lanes).
// K/V tiles for each 32-key step are fetched by the Tensor Data Mover into
// double-buffered LDS; wave 0 issues the descriptors and waits on TENSORcnt.
// ---------------------------------------------------------------------------
__global__ __launch_bounds__(256) void k_attention(const __bf16* __restrict__ qb,
                                                   const __bf16* __restrict__ kall,
                                                   const __bf16* __restrict__ vT,
                                                   __bf16* __restrict__ attn_out) {
    __shared__ __bf16 sK[2][32 * HEAD_DIM];   // [key][d]
    __shared__ __bf16 sV[2][HEAD_DIM * 32];   // [d][key]

    const int lane = threadIdx.x & 31;
    const int wave = threadIdx.x >> 5;
    const int h    = blockIdx.x;
    const int qt   = blockIdx.y * 8 + wave;            // 16-query tile index
    const bool valid = (qt * 16 < SEQ);
    const int qtc  = valid ? qt : 0;

    const int l16 = lane & 15;
    const int kh  = (lane >> 4) << 3;                  // 0 or 8

    const __bf16* kbase = kall + (size_t)h * CACHE_SIZE * HEAD_DIM;
    const __bf16* vbase = vT   + (size_t)h * HEAD_DIM * CACHE_SIZE;

    // Q^T B-fragments: lane -> query qt*16+l16, elements -> d chunk (c*32 + kh..)
    Frag qf[4];
    const __bf16* qp = qb + ((size_t)h * SEQ + qtc * 16 + l16) * HEAD_DIM;
#pragma unroll
    for (int c = 0; c < 4; ++c) {
        qf[c].h[0] = *(const v8bf*)(qp + c * 32 + kh);
        qf[c].h[1] = *(const v8bf*)(qp + c * 32 + kh + 16);
    }

    const unsigned int ldsK[2] = { (unsigned int)(__UINTPTR_TYPE__)&sK[0][0],
                                   (unsigned int)(__UINTPTR_TYPE__)&sK[1][0] };
    const unsigned int ldsV[2] = { (unsigned int)(__UINTPTR_TYPE__)&sV[0][0],
                                   (unsigned int)(__UINTPTR_TYPE__)&sV[1][0] };

    // Prologue: stage tiles for kb=0 into buffer 0.
    if (wave == 0) {
        // K tile: 32 contiguous rows of 128 bf16 -> one dense 1-D copy (8 KB)
        tdm_load_2d_bf16(ldsK[0], kbase, 32 * HEAD_DIM, 1, 32 * HEAD_DIM, 1,
                         (unsigned long long)(32 * HEAD_DIM));
        // V^T tile: 128 rows x 32 keys, row stride CACHE_SIZE -> packed [d][32]
        tdm_load_2d_bf16(ldsV[0], vbase, 32, HEAD_DIM, 32, HEAD_DIM,
                         (unsigned long long)CACHE_SIZE);
        __builtin_amdgcn_s_wait_tensorcnt(0);
    }
    __syncthreads();

    v8f acc[8] = {};                                   // out^T: d = 16t + r + kh
    float mrow = -1e30f;
    float lrow = 0.f;
    int p = 0;

    for (int kb = 0; kb < CACHE_SIZE; kb += 32, p ^= 1) {
        // Prefetch next tile into the other buffer while this one is consumed.
        if (wave == 0 && (kb + 32) < CACHE_SIZE) {
            tdm_load_2d_bf16(ldsK[p ^ 1], kbase + (size_t)(kb + 32) * HEAD_DIM,
                             32 * HEAD_DIM, 1, 32 * HEAD_DIM, 1,
                             (unsigned long long)(32 * HEAD_DIM));
            tdm_load_2d_bf16(ldsV[p ^ 1], vbase + (kb + 32), 32, HEAD_DIM, 32, HEAD_DIM,
                             (unsigned long long)CACHE_SIZE);
        }

        const __bf16* cK = sK[p];
        const __bf16* cV = sV[p];

        // --- S^T = K * Q^T  (two 16-key tiles) ---
        v8f s0 = {}, s1 = {};
#pragma unroll
        for (int c = 0; c < 4; ++c) {
            const __bf16* kp0 = cK + (size_t)l16 * HEAD_DIM + c * 32 + kh;
            const __bf16* kp1 = cK + (size_t)(l16 + 16) * HEAD_DIM + c * 32 + kh;
            Frag ka, kb2;
            ka.h[0]  = *(const v8bf*)kp0;  ka.h[1]  = *(const v8bf*)(kp0 + 16);
            kb2.h[0] = *(const v8bf*)kp1;  kb2.h[1] = *(const v8bf*)(kp1 + 16);
            s0 = wmma_bf16(ka.f,  qf[c].f, s0);
            s1 = wmma_bf16(kb2.f, qf[c].f, s1);
        }

        // --- online softmax (query per lane; keys in VGPR index) ---
        float bm = s0[0];
#pragma unroll
        for (int r = 0; r < 8; ++r) { bm = fmaxf(bm, s0[r]); bm = fmaxf(bm, s1[r]); }
        bm = fmaxf(bm, __shfl_xor(bm, 16, 32));
        const float mnew = fmaxf(mrow, bm);
        const float corr = __expf(mrow - mnew);

        Frag pf;
        float psum = 0.f;
#pragma unroll
        for (int r = 0; r < 8; ++r) {
            const float p0 = __expf(s0[r] - mnew);
            const float p1 = __expf(s1[r] - mnew);
            psum += p0 + p1;
            pf.e[r]     = (__bf16)p0;   // key = kh + r       (tile 0)
            pf.e[r + 8] = (__bf16)p1;   // key = 16 + kh + r  (tile 1)
        }
        psum += __shfl_xor(psum, 16, 32);
        lrow = lrow * corr + psum;
        mrow = mnew;

        // --- out^T += V^T * P^T ---
#pragma unroll
        for (int t = 0; t < 8; ++t) {
#pragma unroll
            for (int r = 0; r < 8; ++r) acc[t][r] *= corr;
            const __bf16* vp = cV + (size_t)(16 * t + l16) * 32 + kh;
            Frag va;
            va.h[0] = *(const v8bf*)vp;
            va.h[1] = *(const v8bf*)(vp + 16);
            acc[t] = wmma_bf16(va.f, pf.f, acc[t]);
        }

        // Next buffer must be fully written before anyone reads it.
        if (wave == 0) __builtin_amdgcn_s_wait_tensorcnt(0);
        __syncthreads();
    }

    if (valid) {
        const float inv = 1.0f / lrow;
        const int q = qt * 16 + l16;
#pragma unroll
        for (int t = 0; t < 8; ++t) {
            Out8 o;
#pragma unroll
            for (int r = 0; r < 8; ++r) o.e[r] = (__bf16)(acc[t][r] * inv);
            __bf16* dst = attn_out + (size_t)q * DIM + h * HEAD_DIM + 16 * t + kh;
            *(v8bf*)dst = o.v;
        }
    }
}

// ---------------------------------------------------------------------------
// Host launch
// ---------------------------------------------------------------------------
extern "C" void kernel_launch(void* const* d_in, const int* in_sizes, int n_in,
                              void* d_out, int out_size, void* d_ws, size_t ws_size,
                              hipStream_t stream) {
    const float* x        = (const float*)d_in[0];
    const float* q_w      = (const float*)d_in[1];
    const float* q_b      = (const float*)d_in[2];
    const float* k_w      = (const float*)d_in[3];
    const float* k_b      = (const float*)d_in[4];
    const float* v_w      = (const float*)d_in[5];
    const float* v_b      = (const float*)d_in[6];
    const float* o_w      = (const float*)d_in[7];
    const float* o_b      = (const float*)d_in[8];
    const float* norm_q_w = (const float*)d_in[9];
    const float* norm_k_w = (const float*)d_in[10];
    const float* cache_k  = (const float*)d_in[11];
    const float* cache_v  = (const float*)d_in[12];
    const float* fcos     = (const float*)d_in[13];
    const float* fsin     = (const float*)d_in[14];
    const int*   cur      = (const int*)d_in[15];
    const int*   sink     = (const int*)d_in[17];
    float* out            = (float*)d_out;

    // workspace carve-up
    char* w = (char*)d_ws;
    auto alloc = [&](size_t bytes) {
        void* p = (void*)w;
        w += (bytes + 255) & ~(size_t)255;
        return p;
    };
    const size_t wbytes = (size_t)DIM * DIM * sizeof(__bf16);
    __bf16* wq   = (__bf16*)alloc(wbytes);
    __bf16* wk   = (__bf16*)alloc(wbytes);
    __bf16* wv   = (__bf16*)alloc(wbytes);
    __bf16* wo   = (__bf16*)alloc(wbytes);
    __bf16* xb   = (__bf16*)alloc((size_t)SEQ * DIM * sizeof(__bf16));
    float*  yq   = (float*) alloc((size_t)SEQ * DIM * sizeof(float));
    float*  yk   = (float*) alloc((size_t)SEQ * DIM * sizeof(float));
    float*  yv   = (float*) alloc((size_t)SEQ * DIM * sizeof(float));
    __bf16* qb   = (__bf16*)alloc((size_t)N_HEADS * SEQ * HEAD_DIM * sizeof(__bf16));
    __bf16* kall = (__bf16*)alloc((size_t)N_HEADS * CACHE_SIZE * HEAD_DIM * sizeof(__bf16));
    __bf16* vT   = (__bf16*)alloc((size_t)N_HEADS * CACHE_SIZE * HEAD_DIM * sizeof(__bf16));
    __bf16* attn = (__bf16*)alloc((size_t)SEQ * DIM * sizeof(__bf16));

    const int nW = DIM * DIM;
    const int nX = SEQ * DIM;
    k_f32_to_bf16<<<(nW + 255) / 256, 256, 0, stream>>>(q_w, wq, nW);
    k_f32_to_bf16<<<(nW + 255) / 256, 256, 0, stream>>>(k_w, wk, nW);
    k_f32_to_bf16<<<(nW + 255) / 256, 256, 0, stream>>>(v_w, wv, nW);
    k_f32_to_bf16<<<(nW + 255) / 256, 256, 0, stream>>>(o_w, wo, nW);
    k_f32_to_bf16<<<(nX + 255) / 256, 256, 0, stream>>>(x, xb, nX);

    // QKV projections: M=720, N=2048, K=2048; 45*32 waves -> 180 blocks
    const int tilesN = DIM / 64;
    const int gemmBlocks = (45 * tilesN + 7) / 8;
    k_gemm_nt<<<gemmBlocks, 256, 0, stream>>>(xb, wq, q_b, yq, SEQ, DIM, DIM, tilesN);
    k_gemm_nt<<<gemmBlocks, 256, 0, stream>>>(xb, wk, k_b, yk, SEQ, DIM, DIM, tilesN);
    k_gemm_nt<<<gemmBlocks, 256, 0, stream>>>(xb, wv, v_b, yv, SEQ, DIM, DIM, tilesN);

    // cache conversion (before the epilogues that overwrite the current window)
    const int nC = CACHE_SIZE * N_HEADS * HEAD_DIM;
    k_cache_convert<<<(nC + 255) / 256, 256, 0, stream>>>(cache_k, cache_v, kall, vT);

    const float smscale = 0.08838834764831845f;  // 1/sqrt(128), folded into Q
    k_norm_rope<<<SEQ, 256, 0, stream>>>(yq, norm_q_w, fcos, fsin, qb, SEQ,
                                         nullptr, nullptr, smscale);
    k_norm_rope<<<SEQ, 256, 0, stream>>>(yk, norm_k_w, fcos, fsin, kall, CACHE_SIZE,
                                         cur, sink, 1.0f);
    k_v_transpose<<<(nX + 255) / 256, 256, 0, stream>>>(yv, vT, cur, sink);

    // flash attention: 16 heads x 6 blocks of 128 queries
    k_attention<<<dim3(N_HEADS, 6), 256, 0, stream>>>(qb, kall, vT, attn);

    // output projection -> f32 d_out
    k_gemm_nt<<<gemmBlocks, 256, 0, stream>>>(attn, wo, o_b, out, SEQ, DIM, DIM, tilesN);

    (void)in_sizes; (void)n_in; (void)out_size; (void)ws_size;
}
